// CrossAttention_2705829396849
// MI455X (gfx1250) — compile-verified, matrix operand order checked by reading, not run on hardware
//
#include <hip/hip_runtime.h>

typedef __attribute__((ext_vector_type(16))) _Float16 v16h;
typedef __attribute__((ext_vector_type(8)))  _Float16 v8h;
typedef __attribute__((ext_vector_type(8)))  float    v8f;

#define WMMA_F32_F16(a, b, c) \
  __builtin_amdgcn_wmma_f32_16x16x32_f16(false, (a), false, (b), (short)0, (c), false, false)

// Tensor Data Mover path. Both toolchains declare the builtin, with different
// arity (probe-confirmed): ROCm 7.2 / clang-22 -> 5 args; amdgpu-toolchain /
// clang-23 -> 6 args (extra int32x8 before cpol). Select by clang major.
#if defined(__HIP_DEVICE_COMPILE__)
#if __has_builtin(__builtin_amdgcn_tensor_load_to_lds)
#define USE_TDM 1
#if (__clang_major__ >= 23)
#define TDM_SIX_ARGS 1
#else
#define TDM_SIX_ARGS 0
#endif
#endif
#endif
#ifndef USE_TDM
#define USE_TDM 0
#endif

// ---------------------------------------------------------------------------
// Problem constants
// ---------------------------------------------------------------------------
constexpr int BSZ   = 4;
constexpr int SEQN  = 2048;
constexpr int SEQM  = 2048;
constexpr int DIM   = 1024;
constexpr int HEADS = 16;
constexpr int DH    = 64;
constexpr int INNER = 1024;
constexpr int KVLEN = SEQM + 1;     // 2049 (null token prepended)
constexpr int KVPAD = 2080;         // 65 * 32
constexpr int ROWS  = BSZ * SEQN;   // 8192
constexpr int KSTR  = 80;           // padded LDS row stride for K tiles (f16)

__device__ __forceinline__ v8f zero_v8f() {
  v8f z = {0.f, 0.f, 0.f, 0.f, 0.f, 0.f, 0.f, 0.f};
  return z;
}

// 16-element f16 fragment load: elements 0..7 from p[0..7], 8..15 from p[16..23].
// With p = row_base + (lane>>4)*8 this matches the ISA 16-bit A/B fragment layout.
__device__ __forceinline__ v16h frag_ld(const _Float16* p) {
  v8h lo = *(const v8h*)(p);
  v8h hi = *(const v8h*)(p + 16);
  v16h a;
#pragma unroll
  for (int i = 0; i < 8; ++i) { a[i] = lo[i]; a[i + 8] = hi[i]; }
  return a;
}

#if USE_TDM
typedef __attribute__((ext_vector_type(4))) unsigned int u32x4;
typedef __attribute__((ext_vector_type(8))) int          i32x8;
typedef __attribute__((ext_vector_type(4))) int          i32x4;

// DMA a 32x64 f16 tile (row-major, row stride 64) from global into LDS,
// padding each 128B row with 32B so the LDS layout has stride KSTR=80 f16.
__device__ __forceinline__ void tdm_load_k_tile(const _Float16* gsrc,
                                                unsigned int lds_byte_addr) {
  const unsigned long long ga = (unsigned long long)(uintptr_t)gsrc;
  u32x4 g0;
  g0[0] = 1u;                                   // count=1 user descriptor
  g0[1] = lds_byte_addr;                        // lds_addr
  g0[2] = (unsigned int)ga;                     // global_addr[31:0]
  g0[3] = (unsigned int)((ga >> 32) & 0x01FFFFFFu) | 0x80000000u;  // [56:32]|type=2
  i32x8 g1;
  g1[0] = (1 << 16)      // data_size = 1 -> 2 bytes
        | (1 << 20)      // pad_enable
        | (4 << 22)      // pad_interval: 32 DWORDs (one 64-f16 row)
        | (7 << 25);     // pad_amount: 8 DWORDs (16 f16) -> stride 80
  g1[1] = (64 << 16);    // tensor_dim0[15:0] = 64 (bits 63:48)
  g1[2] = (32 << 16);    // tensor_dim0[31:16]=0 ; tensor_dim1[15:0] = 32
  g1[3] = (64 << 16);    // tensor_dim1[31:16]=0 ; tile_dim0 = 64
  g1[4] = 32;            // tile_dim1 = 32 ; tile_dim2 = 0
  g1[5] = 64;            // tensor_dim0_stride[31:0] = 64
  g1[6] = 0;             // stride0[47:32]=0 ; tensor_dim1_stride lo = 0
  g1[7] = 0;
  i32x4 gz;
  gz[0] = 0; gz[1] = 0; gz[2] = 0; gz[3] = 0;
#if TDM_SIX_ARGS
  i32x8 gz8;
#pragma unroll
  for (int i = 0; i < 8; ++i) gz8[i] = 0;
  __builtin_amdgcn_tensor_load_to_lds(g0, g1, gz, gz, gz8, 0);
#else
  __builtin_amdgcn_tensor_load_to_lds(g0, g1, gz, gz, 0);
#endif
}

__device__ __forceinline__ void tdm_wait() {
#if __has_builtin(__builtin_amdgcn_s_wait_tensorcnt)
  __builtin_amdgcn_s_wait_tensorcnt(0);
#else
  asm volatile("s_wait_tensorcnt 0x0" ::: "memory");
#endif
}
#endif  // USE_TDM

// ---------------------------------------------------------------------------
// LayerNorm over 1024 cols, one row per block
// ---------------------------------------------------------------------------
template <typename OutT>
__global__ void __launch_bounds__(256) ln_kernel(const float* __restrict__ x,
                                                 const float* __restrict__ g,
                                                 OutT* __restrict__ out) {
  const size_t row = blockIdx.x;
  const float* xr = x + row * DIM;
  float v[4];
  float s = 0.f;
#pragma unroll
  for (int i = 0; i < 4; ++i) { v[i] = xr[threadIdx.x + 256 * i]; s += v[i]; }
  __shared__ float red[8];
#pragma unroll
  for (int msk = 16; msk >= 1; msk >>= 1) s += __shfl_xor(s, msk, 32);
  if ((threadIdx.x & 31) == 0) red[threadIdx.x >> 5] = s;
  __syncthreads();
  float tot = 0.f;
#pragma unroll
  for (int w = 0; w < 8; ++w) tot += red[w];
  const float mean = tot * (1.0f / (float)DIM);
  float sq = 0.f;
#pragma unroll
  for (int i = 0; i < 4; ++i) { const float d = v[i] - mean; sq += d * d; }
#pragma unroll
  for (int msk = 16; msk >= 1; msk >>= 1) sq += __shfl_xor(sq, msk, 32);
  __syncthreads();
  if ((threadIdx.x & 31) == 0) red[threadIdx.x >> 5] = sq;
  __syncthreads();
  float tots = 0.f;
#pragma unroll
  for (int w = 0; w < 8; ++w) tots += red[w];
  const float rstd = rsqrtf(tots * (1.0f / (float)DIM) + 1e-5f);
#pragma unroll
  for (int i = 0; i < 4; ++i) {
    const int c = threadIdx.x + 256 * i;
    out[row * DIM + c] = (OutT)((v[i] - mean) * rstd * g[c]);
  }
}

// ---------------------------------------------------------------------------
// fp32 -> f16 elementwise
// ---------------------------------------------------------------------------
__global__ void cvt_f16_kernel(const float* __restrict__ in,
                               _Float16* __restrict__ out, int n) {
  for (int i = blockIdx.x * blockDim.x + threadIdx.x; i < n;
       i += blockDim.x * gridDim.x)
    out[i] = (_Float16)in[i];
}

// ---------------------------------------------------------------------------
// Softmax bound: sim bounded by 8 * max_d |q_scale[d] * k_scale[d]|
// (q, k are l2-normalized -> Cauchy-Schwarz).
// ---------------------------------------------------------------------------
__global__ void bound_kernel(const float* __restrict__ q_scale,
                             const float* __restrict__ k_scale,
                             float* __restrict__ gb) {
  __shared__ float red[64];
  const int t = threadIdx.x;  // 64 threads
  red[t] = fabsf(q_scale[t] * k_scale[t]);
  __syncthreads();
  if (t == 0) {
    float m = 0.f;
#pragma unroll 8
    for (int i = 0; i < 64; ++i) m = fmaxf(m, red[i]);
    gb[0] = 8.0f * m;
  }
}

// ---------------------------------------------------------------------------
// Null-kv token (row 0, k l2-normalized * k_scale) + zero pad rows
// ---------------------------------------------------------------------------
__global__ void kv_init_kernel(const float* __restrict__ null_kv,
                               const float* __restrict__ k_scale,
                               _Float16* __restrict__ K,
                               _Float16* __restrict__ V) {
  const int bh = blockIdx.x;
  const int d  = threadIdx.x;  // 64 threads
  float ss = 0.f;
#pragma unroll 8
  for (int i = 0; i < DH; ++i) ss += null_kv[i] * null_kv[i];
  const float inv = 1.0f / fmaxf(sqrtf(ss), 1e-12f);
  const size_t base = (size_t)bh * KVPAD * DH;
  K[base + d] = (_Float16)(null_kv[d] * inv * k_scale[d]);
  V[base + d] = (_Float16)(null_kv[DH + d]);
  for (int r = KVLEN; r < KVPAD; ++r) {
    K[base + (size_t)r * DH + d] = (_Float16)0.0f;
    V[base + (size_t)r * DH + d] = (_Float16)0.0f;
  }
}

// ---------------------------------------------------------------------------
// WMMA GEMM with double-buffered LDS B tile.
// MODE 0: Q proj -> l2norm * q_scale * SCALE -> f16
// MODE 1: KV proj -> K: l2norm * k_scale ; V raw -> f16
// MODE 2: Out proj -> raw fp32
// ---------------------------------------------------------------------------
template <int MODE>
__global__ void __launch_bounds__(256) gemm_wmma_kernel(
    const _Float16* __restrict__ A, const _Float16* __restrict__ Bw, int ncols,
    const float* __restrict__ svec, _Float16* __restrict__ outA,
    _Float16* __restrict__ outB, float* __restrict__ out32) {
  __shared__ alignas(16) _Float16 Blds[2][64 * 40];
  const int tid   = threadIdx.x;
  const int wave  = tid >> 5;
  const int lane  = tid & 31;
  const int mrow  = lane & 15;
  const int khalf = lane >> 4;
  const int c0    = blockIdx.x * 64;
  const int row0  = blockIdx.y * 128 + wave * 16;
  const int ck    = tid >> 3;
  const int cn    = (tid & 7) * 8;

  const _Float16* Arow = A + (size_t)(row0 + mrow) * DIM + khalf * 8;

  v8f acc[4];
#pragma unroll
  for (int t = 0; t < 4; ++t) acc[t] = zero_v8f();

  {  // stage first B tile (transposed: Blds[n][k], stride 40)
    v8h vb = *(const v8h*)(Bw + (size_t)ck * ncols + c0 + cn);
#pragma unroll
    for (int i = 0; i < 8; ++i) Blds[0][(cn + i) * 40 + ck] = vb[i];
  }
  __syncthreads();

  for (int k0 = 0; k0 < DIM; k0 += 32) {
    const int  cur     = (k0 >> 5) & 1;
    const bool hasNext = (k0 + 32) < DIM;
    v8h vbn;
    if (hasNext)
      vbn = *(const v8h*)(Bw + (size_t)(k0 + 32 + ck) * ncols + c0 + cn);
    const v16h af = frag_ld(Arow + k0);
#pragma unroll
    for (int t = 0; t < 4; ++t) {
      const v16h bf = frag_ld(&Blds[cur][(t * 16 + mrow) * 40 + khalf * 8]);
      acc[t] = WMMA_F32_F16(af, bf, acc[t]);
    }
    if (hasNext) {
#pragma unroll
      for (int i = 0; i < 8; ++i) Blds[cur ^ 1][(cn + i) * 40 + ck] = vbn[i];
    }
    __syncthreads();
  }

  if (MODE == 2) {
#pragma unroll
    for (int t = 0; t < 4; ++t)
#pragma unroll
      for (int r = 0; r < 8; ++r)
        out32[(size_t)(row0 + khalf * 8 + r) * DIM + c0 + t * 16 + mrow] = acc[t][r];
    return;
  }

  const bool isK = (MODE == 0) || (c0 < INNER);
  float inv[8];
#pragma unroll
  for (int r = 0; r < 8; ++r) inv[r] = 1.0f;
  if (isK) {
#pragma unroll
    for (int r = 0; r < 8; ++r) {
      float ss = 0.f;
#pragma unroll
      for (int t = 0; t < 4; ++t) ss += acc[t][r] * acc[t][r];
#pragma unroll
      for (int msk = 1; msk < 16; msk <<= 1) ss += __shfl_xor(ss, msk, 32);
      inv[r] = 1.0f / fmaxf(sqrtf(ss), 1e-12f);
    }
  }
  const int hcol = (MODE == 0) ? (c0 / DH) : ((c0 < INNER ? c0 : c0 - INNER) / DH);
#pragma unroll
  for (int t = 0; t < 4; ++t) {
#pragma unroll
    for (int r = 0; r < 8; ++r) {
      const int grow = row0 + khalf * 8 + r;
      const int bb   = grow >> 11;
      const int seq  = grow & 2047;
      const int d    = t * 16 + mrow;
      if (MODE == 0) {
        const float val = acc[t][r] * inv[r] * svec[d] * 8.0f;
        outA[(((size_t)(bb * HEADS + hcol)) * SEQN + seq) * DH + d] = (_Float16)val;
      } else {
        if (isK) {
          const float val = acc[t][r] * inv[r] * svec[d];
          outA[(((size_t)(bb * HEADS + hcol)) * KVPAD + 1 + seq) * DH + d] = (_Float16)val;
        } else {
          outB[(((size_t)(bb * HEADS + hcol)) * KVPAD + 1 + seq) * DH + d] =
              (_Float16)acc[t][r];
        }
      }
    }
  }
}

// ---------------------------------------------------------------------------
// Fused attention, single-pass bounded softmax.
// grid = (N/128, HEADS, B), block = 256 (8 waves); wave -> 16 query rows.
// K tile (shared by all 8 waves) staged block-wide into LDS via the Tensor
// Data Mover (double-buffered, issued by wave 0, padded to stride-80 rows);
// cooperative-copy fallback when the TDM builtin is unavailable.
// Per 32-key step: 4 QK^T + 4 PV + 1 rowsum WMMA (all-ones B fragment).
// ---------------------------------------------------------------------------
__global__ void __launch_bounds__(256) attn_kernel(
    const _Float16* __restrict__ Q, const _Float16* __restrict__ K,
    const _Float16* __restrict__ V, const float* __restrict__ gb,
    _Float16* __restrict__ AO) {
  __shared__ alignas(16) _Float16 Klds[2][32 * KSTR];
  __shared__ alignas(16) _Float16 Vlds[2][64 * 40];
  __shared__ alignas(16) _Float16 Plds[8][16 * 40];
  const int tid   = threadIdx.x;
  const int wave  = tid >> 5;
  const int lane  = tid & 31;
  const int mrow  = lane & 15;
  const int khalf = lane >> 4;
  const int b  = blockIdx.z;
  const int h  = blockIdx.y;
  const int n0 = blockIdx.x * 128 + wave * 16;
  const int ck = tid >> 3;         // tile row 0..31
  const int cn = (tid & 7) * 8;    // 8-elem chunk within 64-elem row

  const float gbound = gb[0];
  const size_t qbase  = (((size_t)(b * HEADS + h)) * SEQN + n0) * DH;
  const size_t kvbase = ((size_t)(b * HEADS + h)) * KVPAD * DH;

  const _Float16* qp = Q + qbase + (size_t)mrow * DH + khalf * 8;
  const v16h qa0 = frag_ld(qp);        // k = 0..31
  const v16h qa1 = frag_ld(qp + 32);   // k = 32..63

  v16h ones;
#pragma unroll
  for (int i = 0; i < 16; ++i) ones[i] = (_Float16)1.0f;

  v8f o[4];
#pragma unroll
  for (int t = 0; t < 4; ++t) o[t] = zero_v8f();
  v8f lacc = zero_v8f();

  // ---- stage first K tile ----
#if USE_TDM
  if (wave == 0)
    tdm_load_k_tile(K + kvbase, (unsigned int)(uintptr_t)(&Klds[0][0]));
#else
  {
    v8h kk = *(const v8h*)(K + kvbase + (size_t)ck * DH + cn);
    *(v8h*)(&Klds[0][ck * KSTR + cn]) = kk;
  }
#endif
  // ---- stage first V tile (transposed: Vlds[d][j_local], stride 40) ----
  {
    v8h vv = *(const v8h*)(V + kvbase + (size_t)ck * DH + cn);
#pragma unroll
    for (int i = 0; i < 8; ++i) Vlds[0][(cn + i) * 40 + ck] = vv[i];
  }
#if USE_TDM
  if (wave == 0) tdm_wait();
#endif
  __syncthreads();

  for (int j0 = 0; j0 < KVPAD; j0 += 32) {
    const int  cur     = (j0 >> 5) & 1;
    const bool hasNext = (j0 + 32) < KVPAD;
    v8h vvn;
#if !USE_TDM
    v8h kkn;
#endif
    if (hasNext) {
#if USE_TDM
      if (wave == 0)
        tdm_load_k_tile(K + kvbase + (size_t)(j0 + 32) * DH,
                        (unsigned int)(uintptr_t)(&Klds[cur ^ 1][0]));
#else
      kkn = *(const v8h*)(K + kvbase + (size_t)(j0 + 32 + ck) * DH + cn);
#endif
      vvn = *(const v8h*)(V + kvbase + (size_t)(j0 + 32 + ck) * DH + cn);
    }
    _Float16* P = Plds[wave];
    // ---- S = q.k^T from LDS K tile ; p = exp(s - bound) -> P tile ----
#pragma unroll
    for (int st = 0; st < 2; ++st) {
      const int j = j0 + st * 16 + mrow;
      const _Float16* kp = &Klds[cur][(st * 16 + mrow) * KSTR + khalf * 8];
      const v16h kb0 = frag_ld(kp);
      const v16h kb1 = frag_ld(kp + 32);
      v8f sz = zero_v8f();
      sz = WMMA_F32_F16(qa0, kb0, sz);
      sz = WMMA_F32_F16(qa1, kb1, sz);
      const bool valid = (j < KVLEN);
#pragma unroll
      for (int r = 0; r < 8; ++r) {
        const float p = valid ? __expf(sz[r] - gbound) : 0.0f;
        P[(khalf * 8 + r) * 40 + st * 16 + mrow] = (_Float16)p;
      }
    }
    // ---- O += P.V ; rowsum += P.1 ----
    const v16h pa = frag_ld(P + (size_t)mrow * 40 + khalf * 8);
#pragma unroll
    for (int t = 0; t < 4; ++t) {
      const v16h vb = frag_ld(&Vlds[cur][(t * 16 + mrow) * 40 + khalf * 8]);
      o[t] = WMMA_F32_F16(pa, vb, o[t]);
    }
    lacc = WMMA_F32_F16(pa, ones, lacc);
    if (hasNext) {
#if !USE_TDM
      *(v8h*)(&Klds[cur ^ 1][ck * KSTR + cn]) = kkn;
#endif
#pragma unroll
      for (int i = 0; i < 8; ++i) Vlds[cur ^ 1][(cn + i) * 40 + ck] = vvn[i];
    }
#if USE_TDM
    if (hasNext && wave == 0) tdm_wait();
#endif
    __syncthreads();
  }
  // ---- normalize and store to [B*N, INNER] (b, n, h, d) ----
#pragma unroll
  for (int r = 0; r < 8; ++r) {
    const float invl = 1.0f / lacc[r];
#pragma unroll
    for (int t = 0; t < 4; ++t) {
      AO[((size_t)(b * SEQN + n0 + khalf * 8 + r)) * INNER + h * DH + t * 16 + mrow] =
          (_Float16)(o[t][r] * invl);
    }
  }
}

// ---------------------------------------------------------------------------
// Launcher
// ---------------------------------------------------------------------------
extern "C" void kernel_launch(void* const* d_in, const int* in_sizes, int n_in,
                              void* d_out, int out_size, void* d_ws, size_t ws_size,
                              hipStream_t stream) {
  (void)in_sizes; (void)n_in; (void)out_size; (void)ws_size;
  const float* x       = (const float*)d_in[0];
  const float* context = (const float*)d_in[1];
  const float* g_norm  = (const float*)d_in[2];
  const float* null_kv = (const float*)d_in[3];
  const float* Wq      = (const float*)d_in[4];
  const float* Wkv     = (const float*)d_in[5];
  const float* q_scale = (const float*)d_in[6];
  const float* k_scale = (const float*)d_in[7];
  const float* Wout    = (const float*)d_in[8];
  const float* g_out   = (const float*)d_in[9];
  float* out = (float*)d_out;

  char* ws = (char*)d_ws;
  size_t off = 0;
  auto take = [&](size_t bytes) {
    char* p = ws + off;
    off += (bytes + 255) & ~(size_t)255;
    return p;
  };
  _Float16* xn16   = (_Float16*)take((size_t)ROWS * DIM * 2);
  _Float16* ctx16  = (_Float16*)take((size_t)BSZ * SEQM * DIM * 2);
  _Float16* Wq16   = (_Float16*)take((size_t)DIM * INNER * 2);
  _Float16* Wkv16  = (_Float16*)take((size_t)DIM * 2 * INNER * 2);
  _Float16* Wout16 = (_Float16*)take((size_t)INNER * DIM * 2);
  _Float16* Qb     = (_Float16*)take((size_t)BSZ * HEADS * SEQN * DH * 2);
  _Float16* Kb     = (_Float16*)take((size_t)BSZ * HEADS * KVPAD * DH * 2);
  _Float16* Vb     = (_Float16*)take((size_t)BSZ * HEADS * KVPAD * DH * 2);
  _Float16* AO     = (_Float16*)take((size_t)ROWS * INNER * 2);
  float*    tmp    = (float*)take((size_t)ROWS * DIM * 4);
  float*    gb     = (float*)take(sizeof(float));

  // 1. LayerNorm(x) -> f16 ; fp32 -> f16 conversions ; softmax bound
  ln_kernel<_Float16><<<ROWS, 256, 0, stream>>>(x, g_norm, xn16);
  cvt_f16_kernel<<<2048, 256, 0, stream>>>(context, ctx16, BSZ * SEQM * DIM);
  cvt_f16_kernel<<<1024, 256, 0, stream>>>(Wq, Wq16, DIM * INNER);
  cvt_f16_kernel<<<2048, 256, 0, stream>>>(Wkv, Wkv16, DIM * 2 * INNER);
  cvt_f16_kernel<<<1024, 256, 0, stream>>>(Wout, Wout16, INNER * DIM);
  bound_kernel<<<1, 64, 0, stream>>>(q_scale, k_scale, gb);

  // 2. null-kv token + pad rows
  kv_init_kernel<<<BSZ * HEADS, 64, 0, stream>>>(null_kv, k_scale, Kb, Vb);

  // 3. projections (WMMA) with fused l2norm epilogues
  gemm_wmma_kernel<0><<<dim3(INNER / 64, ROWS / 128), 256, 0, stream>>>(
      xn16, Wq16, INNER, q_scale, Qb, nullptr, nullptr);
  gemm_wmma_kernel<1><<<dim3(2 * INNER / 64, ROWS / 128), 256, 0, stream>>>(
      ctx16, Wkv16, 2 * INNER, k_scale, Kb, Vb, nullptr);

  // 4. fused attention (WMMA + TDM-staged K tiles)
  attn_kernel<<<dim3(SEQN / 128, HEADS, BSZ), 256, 0, stream>>>(Qb, Kb, Vb, gb, AO);

  // 5. output projection (WMMA) + final LayerNorm
  gemm_wmma_kernel<2><<<dim3(DIM / 64, ROWS / 128), 256, 0, stream>>>(
      AO, Wout16, DIM, nullptr, nullptr, nullptr, tmp);
  ln_kernel<float><<<8192, 256, 0, stream>>>(tmp, g_out, out);
}
